// ResAGNN_34918084116975
// MI455X (gfx1250) — compile-verified, matrix operand order checked by reading, not run on hardware
//
#include <hip/hip_runtime.h>
#include <hip/hip_bf16.h>

typedef __attribute__((ext_vector_type(16))) __bf16 v16bf;
typedef __attribute__((ext_vector_type(8)))  float  v8f;

#define DWIDTH 67   // residual state width
#define HIDW   64
#define INCH   3

// ---------- bf16 helpers (round-to-nearest-even) ----------
__device__ __forceinline__ unsigned short f2bf(float f) {
  union { float f; unsigned u; } x; x.f = f;
  unsigned u = x.u;
  u += 0x7FFFu + ((u >> 16) & 1u);
  return (unsigned short)(u >> 16);
}
__device__ __forceinline__ float bf2f(unsigned short h) {
  union { unsigned u; float f; } x; x.u = ((unsigned)h) << 16;
  return x.f;
}

// ---------- wave-level GEMM: 16 rows (LDS bf16, stride KP) x W^T (global bf16 [NT*16][KP]) ----------
// Accumulates into NT 16x16 f32 tiles via v_wmma_f32_16x16x32_bf16.
template <int KP, int NT>
__device__ __forceinline__ void wave_gemm(const unsigned short* __restrict__ actw,
                                          const unsigned short* __restrict__ wt,
                                          v8f acc[NT]) {
  const int lane = threadIdx.x & 31;
  const int half = lane >> 4;
  const int lm   = lane & 15;
#pragma unroll
  for (int kc = 0; kc < KP / 32; ++kc) {
    union { unsigned int u[8]; v16bf v; } a;
#pragma unroll
    for (int r = 0; r < 8; ++r) {
      const int kk = kc * 32 + (r >> 2) * 16 + half * 8 + (r & 3) * 2;
      a.u[r] = *(const unsigned int*)(actw + lm * KP + kk);
    }
#pragma unroll
    for (int t = 0; t < NT; ++t) {
      union { unsigned int u[8]; v16bf v; } b;
      const int n = t * 16 + lm;
#pragma unroll
      for (int r = 0; r < 8; ++r) {
        const int kk = kc * 32 + (r >> 2) * 16 + half * 8 + (r & 3) * 2;
        b.u[r] = *(const unsigned int*)(wt + n * KP + kk);
      }
      acc[t] = __builtin_amdgcn_wmma_f32_16x16x32_bf16(
          false, a.v, false, b.v, (short)0, acc[t], false, false);
    }
  }
}

// Spill WMMA accumulators to f32 LDS scratch laid out [16 rows][NP].
template <int NT>
__device__ __forceinline__ void store_acc(float* __restrict__ scrw, int NP, const v8f acc[NT]) {
  const int lane = threadIdx.x & 31;
  const int half = lane >> 4;
  const int lm   = lane & 15;
#pragma unroll
  for (int t = 0; t < NT; ++t)
#pragma unroll
    for (int r = 0; r < 8; ++r)
      scrw[(r + 8 * half) * NP + t * 16 + lm] = acc[t][r];
}

// Per-row bias + LayerNorm + tanh; optional bf16 LDS output (zero-padded to kpn) and f32 global output.
__device__ __forceinline__ void row_ln_tanh(const float* __restrict__ h, int nout,
                                            const float* __restrict__ bias,
                                            const float* __restrict__ g,
                                            const float* __restrict__ be,
                                            unsigned short* actrow, int kpn,
                                            float* goutrow) {
  float mean = 0.f;
  for (int j = 0; j < nout; ++j) mean += h[j] + bias[j];
  mean /= (float)nout;
  float var = 0.f;
  for (int j = 0; j < nout; ++j) { float d = h[j] + bias[j] - mean; var += d * d; }
  var /= (float)nout;
  const float inv = rsqrtf(var + 1e-5f);
  for (int j = 0; j < nout; ++j) {
    const float v = tanhf((h[j] + bias[j] - mean) * inv * g[j] + be[j]);
    if (actrow)  actrow[j]  = f2bf(v);
    if (goutrow) goutrow[j] = v;
  }
  if (actrow)
    for (int j = nout; j < kpn; ++j) actrow[j] = 0;
}

// ---------- weight prep: W[K][N] f32 -> W^T[Npad][Kpad] bf16 (zero padded) ----------
__global__ void prep_wt(const float* __restrict__ W, unsigned short* __restrict__ Wt,
                        int K, int N, int Kpad, int Npad) {
  const int idx = blockIdx.x * blockDim.x + threadIdx.x;
  if (idx >= Kpad * Npad) return;
  const int n = idx / Kpad, k = idx % Kpad;
  const float v = (k < K && n < N) ? W[k * N + n] : 0.f;
  Wt[n * Kpad + k] = f2bf(v);
}

__global__ void zero_f32(float* __restrict__ p, long n) {
  const long i = (long)blockIdx.x * blockDim.x + threadIdx.x;
  if (i < n) p[i] = 0.f;
}

// ---------- input MLP: x0[N][3] -> x[N][67] = [tanh-LN MLP(x0), x0] ----------
__global__ __launch_bounds__(128) void input_pass(
    const float* __restrict__ x0, float* __restrict__ x,
    const unsigned short* wt0, const unsigned short* wt1, const unsigned short* wt2,
    const float* b0, const float* g0, const float* be0,
    const float* b1, const float* g1, const float* be1,
    const float* b2, const float* g2, const float* be2, int NN) {
  __shared__ unsigned short sA[64 * 32];
  __shared__ unsigned short sB[64 * 64];
  __shared__ float sScr[64 * 64];
  const int nb  = blockIdx.x * 64;
  const int tid = threadIdx.x;
  const int wv  = tid >> 5;

  for (int i = tid; i < 64 * 32; i += 128) {
    const int row = i / 32, k = i % 32, node = nb + row;
    const float v = (node < NN && k < INCH) ? x0[node * INCH + k] : 0.f;
    sA[i] = f2bf(v);
  }
  __syncthreads();

  { v8f z = {0,0,0,0,0,0,0,0}; v8f acc[4] = {z,z,z,z};
    wave_gemm<32, 4>(sA + wv * 16 * 32, wt0, acc);
    store_acc<4>(sScr + wv * 16 * 64, 64, acc); }
  __syncthreads();
  if ((tid & 31) < 16) {
    const int row = wv * 16 + (tid & 15);
    row_ln_tanh(sScr + row * 64, HIDW, b0, g0, be0, sB + row * 64, 64, nullptr);
  }
  __syncthreads();

  { v8f z = {0,0,0,0,0,0,0,0}; v8f acc[4] = {z,z,z,z};
    wave_gemm<64, 4>(sB + wv * 16 * 64, wt1, acc);
    store_acc<4>(sScr + wv * 16 * 64, 64, acc); }
  __syncthreads();
  if ((tid & 31) < 16) {
    const int row = wv * 16 + (tid & 15);
    row_ln_tanh(sScr + row * 64, HIDW, b1, g1, be1, sB + row * 64, 64, nullptr);
  }
  __syncthreads();

  { v8f z = {0,0,0,0,0,0,0,0}; v8f acc[4] = {z,z,z,z};
    wave_gemm<64, 4>(sB + wv * 16 * 64, wt2, acc);
    store_acc<4>(sScr + wv * 16 * 64, 64, acc); }
  __syncthreads();
  if ((tid & 31) < 16) {
    const int row  = wv * 16 + (tid & 15);
    const int node = nb + row;
    if (node < NN) {
      row_ln_tanh(sScr + row * 64, HIDW, b2, g2, be2, nullptr, 0, x + (long)node * DWIDTH);
      for (int c = 0; c < INCH; ++c)
        x[(long)node * DWIDTH + HIDW + c] = x0[node * INCH + c];
    }
  }
}

// ---------- edge pass: gather -> edge MLP (WMMA) -> sigmoid gate -> atomic scatter (mode 0)
//            or final logits to out (mode 1) ----------
__global__ __launch_bounds__(128) void edge_pass(
    const float* __restrict__ x, const int* __restrict__ eidx,
    const unsigned short* wt0, const unsigned short* wt1, const unsigned short* wt2,
    const float* b0, const float* g0, const float* be0,
    const float* b1, const float* g1, const float* be1,
    const float* b2, const float* g2, const float* be2,
    const float* w3, const float* b3,
    float* __restrict__ msgs, float* __restrict__ out, int E, int mode) {
  __shared__ unsigned short sA[64 * 160];
  __shared__ unsigned short sB[64 * 96];
  __shared__ float sScr[64 * 80];
  __shared__ int   sSrc[64], sDst[64];
  __shared__ float sE[64];
  const int eb  = blockIdx.x * 64;
  const int tid = threadIdx.x;
  const int wv  = tid >> 5;
  const int* startp = eidx;
  const int* endp   = eidx + E;

  if (tid < 64) {
    const int id = eb + tid;
    sSrc[tid] = (id < E) ? startp[id] : -1;
    sDst[tid] = (id < E) ? endp[id]   : -1;
  }
  __syncthreads();

  for (int i = tid; i < 64 * 160; i += 128) {
    const int row = i / 160, k = i % 160;
    float v = 0.f;
    if (k < DWIDTH)            { const int n = sSrc[row]; if (n >= 0) v = x[(long)n * DWIDTH + k]; }
    else if (k < 2 * DWIDTH)   { const int n = sDst[row]; if (n >= 0) v = x[(long)n * DWIDTH + (k - DWIDTH)]; }
    sA[i] = f2bf(v);
  }
  __syncthreads();

  { v8f z = {0,0,0,0,0,0,0,0}; v8f acc[5] = {z,z,z,z,z};
    wave_gemm<160, 5>(sA + wv * 16 * 160, wt0, acc);
    store_acc<5>(sScr + wv * 16 * 80, 80, acc); }
  __syncthreads();
  if ((tid & 31) < 16) {
    const int row = wv * 16 + (tid & 15);
    row_ln_tanh(sScr + row * 80, DWIDTH, b0, g0, be0, sB + row * 96, 96, nullptr);
  }
  __syncthreads();

  { v8f z = {0,0,0,0,0,0,0,0}; v8f acc[5] = {z,z,z,z,z};
    wave_gemm<96, 5>(sB + wv * 16 * 96, wt1, acc);
    store_acc<5>(sScr + wv * 16 * 80, 80, acc); }
  __syncthreads();
  if ((tid & 31) < 16) {
    const int row = wv * 16 + (tid & 15);
    row_ln_tanh(sScr + row * 80, DWIDTH, b1, g1, be1, sB + row * 96, 96, nullptr);
  }
  __syncthreads();

  { v8f z = {0,0,0,0,0,0,0,0}; v8f acc[5] = {z,z,z,z,z};
    wave_gemm<96, 5>(sB + wv * 16 * 96, wt2, acc);
    store_acc<5>(sScr + wv * 16 * 80, 80, acc); }
  __syncthreads();
  if ((tid & 31) < 16) {
    const int row = wv * 16 + (tid & 15);
    row_ln_tanh(sScr + row * 80, DWIDTH, b2, g2, be2, sB + row * 96, 96, nullptr);
  }
  __syncthreads();

  // final 67 -> 1 linear
  if ((tid & 31) < 16) {
    const int row = wv * 16 + (tid & 15);
    const int id  = eb + row;
    float s = b3[0];
    for (int j = 0; j < DWIDTH; ++j) s += bf2f(sB[row * 96 + j]) * w3[j];
    if (mode == 0) {
      sE[row] = (id < E) ? (1.f / (1.f + expf(-s))) : 0.f;
    } else if (id < E) {
      out[id] = s;
    }
  }
  __syncthreads();

  if (mode == 0) {
    // thread pair per edge: side 0 adds e*x[start] into msgs[end], side 1 adds e*x[end] into msgs[start]
    const int slot = tid >> 1, side = tid & 1;
    const int id = eb + slot;
    if (id < E) {
      const float ev = sE[slot];
      const int src = side ? sDst[slot] : sSrc[slot];
      const int dst = side ? sSrc[slot] : sDst[slot];
      for (int k = 0; k < DWIDTH; ++k)
        atomicAdd(&msgs[(long)dst * DWIDTH + k], ev * x[(long)src * DWIDTH + k]);
    }
  }
}

// ---------- node pass: [msgs, x] -> node MLP -> residual update of x ----------
__global__ __launch_bounds__(128) void node_pass(
    float* __restrict__ x, const float* __restrict__ x0, const float* __restrict__ msgs,
    const unsigned short* wt0, const unsigned short* wt1, const unsigned short* wt2,
    const float* b0, const float* g0, const float* be0,
    const float* b1, const float* g1, const float* be1,
    const float* b2, int NN) {
  __shared__ unsigned short sA[64 * 160];
  __shared__ unsigned short sB[64 * 64];
  __shared__ float sScr[64 * 64];
  const int nb  = blockIdx.x * 64;
  const int tid = threadIdx.x;
  const int wv  = tid >> 5;

  for (int i = tid; i < 64 * 160; i += 128) {
    const int row = i / 160, k = i % 160, node = nb + row;
    float v = 0.f;
    if (node < NN) {
      if (k < DWIDTH)          v = msgs[(long)node * DWIDTH + k];
      else if (k < 2 * DWIDTH) v = x[(long)node * DWIDTH + (k - DWIDTH)];
    }
    sA[i] = f2bf(v);
  }
  __syncthreads();

  { v8f z = {0,0,0,0,0,0,0,0}; v8f acc[4] = {z,z,z,z};
    wave_gemm<160, 4>(sA + wv * 16 * 160, wt0, acc);
    store_acc<4>(sScr + wv * 16 * 64, 64, acc); }
  __syncthreads();
  if ((tid & 31) < 16) {
    const int row = wv * 16 + (tid & 15);
    row_ln_tanh(sScr + row * 64, HIDW, b0, g0, be0, sB + row * 64, 64, nullptr);
  }
  __syncthreads();

  { v8f z = {0,0,0,0,0,0,0,0}; v8f acc[4] = {z,z,z,z};
    wave_gemm<64, 4>(sB + wv * 16 * 64, wt1, acc);
    store_acc<4>(sScr + wv * 16 * 64, 64, acc); }
  __syncthreads();
  if ((tid & 31) < 16) {
    const int row = wv * 16 + (tid & 15);
    row_ln_tanh(sScr + row * 64, HIDW, b1, g1, be1, sB + row * 64, 64, nullptr);
  }
  __syncthreads();

  { v8f z = {0,0,0,0,0,0,0,0}; v8f acc[4] = {z,z,z,z};
    wave_gemm<64, 4>(sB + wv * 16 * 64, wt2, acc);
    store_acc<4>(sScr + wv * 16 * 64, 64, acc); }
  __syncthreads();
  if ((tid & 31) < 16) {
    const int row  = wv * 16 + (tid & 15);
    const int node = nb + row;
    if (node < NN) {
      const float* h = sScr + row * 64;
      for (int j = 0; j < HIDW; ++j)
        x[(long)node * DWIDTH + j] += h[j] + b2[j];
      for (int c = 0; c < INCH; ++c)
        x[(long)node * DWIDTH + HIDW + c] += x0[node * INCH + c];
    }
  }
}

// ---------- host orchestration ----------
static inline size_t al256(size_t v) { return (v + 255) & ~(size_t)255; }

extern "C" void kernel_launch(void* const* d_in, const int* in_sizes, int n_in,
                              void* d_out, int out_size, void* d_ws, size_t ws_size,
                              hipStream_t stream) {
  (void)n_in; (void)out_size; (void)ws_size;
  const float* x0   = (const float*)d_in[0];
  const int*   eidx = (const int*)d_in[1];
  const int NN = in_sizes[0] / INCH;
  const int E  = in_sizes[1] / 2;

  // input_mlp: Ws[0..2], bs[0..2], gs[0..2], betas[0..2]
  const float *iW0 = (const float*)d_in[2],  *iW1 = (const float*)d_in[3],  *iW2 = (const float*)d_in[4];
  const float *ib0 = (const float*)d_in[5],  *ib1 = (const float*)d_in[6],  *ib2 = (const float*)d_in[7];
  const float *ig0 = (const float*)d_in[8],  *ig1 = (const float*)d_in[9],  *ig2 = (const float*)d_in[10];
  const float *ie0 = (const float*)d_in[11], *ie1 = (const float*)d_in[12], *ie2 = (const float*)d_in[13];
  // edge_mlp: Ws[0..3], bs[0..3], gs[0..2], betas[0..2]
  const float *eW0 = (const float*)d_in[14], *eW1 = (const float*)d_in[15], *eW2 = (const float*)d_in[16];
  const float *eW3 = (const float*)d_in[17];
  const float *eb0 = (const float*)d_in[18], *eb1 = (const float*)d_in[19], *eb2 = (const float*)d_in[20];
  const float *eb3 = (const float*)d_in[21];
  const float *eg0 = (const float*)d_in[22], *eg1 = (const float*)d_in[23], *eg2 = (const float*)d_in[24];
  const float *ee0 = (const float*)d_in[25], *ee1 = (const float*)d_in[26], *ee2 = (const float*)d_in[27];
  // node_mlp: Ws[0..2], bs[0..2], gs[0..1], betas[0..1]
  const float *nW0 = (const float*)d_in[28], *nW1 = (const float*)d_in[29], *nW2 = (const float*)d_in[30];
  const float *nb0 = (const float*)d_in[31], *nb1 = (const float*)d_in[32], *nb2 = (const float*)d_in[33];
  const float *ng0 = (const float*)d_in[34], *ng1 = (const float*)d_in[35];
  const float *ne0 = (const float*)d_in[36], *ne1 = (const float*)d_in[37];

  // workspace layout
  char* ws = (char*)d_ws;
  size_t off = 0;
  auto take = [&](size_t bytes) { size_t o = off; off = al256(off + bytes); return o; };
  unsigned short* wtI0 = (unsigned short*)(ws + take(64 * 32  * 2));
  unsigned short* wtI1 = (unsigned short*)(ws + take(64 * 64  * 2));
  unsigned short* wtI2 = (unsigned short*)(ws + take(64 * 64  * 2));
  unsigned short* wtE0 = (unsigned short*)(ws + take(80 * 160 * 2));
  unsigned short* wtE1 = (unsigned short*)(ws + take(80 * 96  * 2));
  unsigned short* wtE2 = (unsigned short*)(ws + take(80 * 96  * 2));
  unsigned short* wtN0 = (unsigned short*)(ws + take(64 * 160 * 2));
  unsigned short* wtN1 = (unsigned short*)(ws + take(64 * 64  * 2));
  unsigned short* wtN2 = (unsigned short*)(ws + take(64 * 64  * 2));
  float* X    = (float*)(ws + take((size_t)NN * DWIDTH * 4));
  float* MSGS = (float*)(ws + take((size_t)NN * DWIDTH * 4));

  // weight prep (bf16 transpose+pad)
  auto prep = [&](const float* W, unsigned short* Wt, int K, int N, int KP, int NP) {
    prep_wt<<<(KP * NP + 255) / 256, 256, 0, stream>>>(W, Wt, K, N, KP, NP);
  };
  prep(iW0, wtI0, INCH, HIDW, 32, 64);
  prep(iW1, wtI1, HIDW, HIDW, 64, 64);
  prep(iW2, wtI2, HIDW, HIDW, 64, 64);
  prep(eW0, wtE0, 2 * DWIDTH, DWIDTH, 160, 80);
  prep(eW1, wtE1, DWIDTH, DWIDTH, 96, 80);
  prep(eW2, wtE2, DWIDTH, DWIDTH, 96, 80);
  prep(nW0, wtN0, 2 * DWIDTH, HIDW, 160, 64);
  prep(nW1, wtN1, HIDW, HIDW, 64, 64);
  prep(nW2, wtN2, HIDW, HIDW, 64, 64);

  const int nodeBlocks = (NN + 63) / 64;
  const int edgeBlocks = (E + 63) / 64;

  input_pass<<<nodeBlocks, 128, 0, stream>>>(x0, X, wtI0, wtI1, wtI2,
                                             ib0, ig0, ie0, ib1, ig1, ie1, ib2, ig2, ie2, NN);

  for (int it = 0; it < 8; ++it) {
    zero_f32<<<((long)NN * DWIDTH + 255) / 256, 256, 0, stream>>>(MSGS, (long)NN * DWIDTH);
    edge_pass<<<edgeBlocks, 128, 0, stream>>>(X, eidx, wtE0, wtE1, wtE2,
                                              eb0, eg0, ee0, eb1, eg1, ee1, eb2, eg2, ee2,
                                              eW3, eb3, MSGS, nullptr, E, 0);
    node_pass<<<nodeBlocks, 128, 0, stream>>>(X, x0, MSGS, wtN0, wtN1, wtN2,
                                              nb0, ng0, ne0, nb1, ng1, ne1, nb2, NN);
  }

  edge_pass<<<edgeBlocks, 128, 0, stream>>>(X, eidx, wtE0, wtE1, wtE2,
                                            eb0, eg0, ee0, eb1, eg1, ee1, eb2, eg2, ee2,
                                            eW3, eb3, nullptr, (float*)d_out, E, 1);
}
